// QMOIPerformerAttention_43327630082344
// MI455X (gfx1250) — matrix operand order repaired and checked by reading, not compiled, not run on hardware
//
#include <hip/hip_runtime.h>
#include <hip/hip_bf16.h>
#include <stdint.h>

// ---------------------------------------------------------------------------
// CDNA5 (gfx1250) Performer attention, bf16 WMMA throughout.
// All matmuls mapped to v_wmma_f32_16x16x32_bf16 (wave32, 16x16 C tiles).
// ---------------------------------------------------------------------------

typedef __attribute__((ext_vector_type(16))) __bf16 v16bf;
typedef __attribute__((ext_vector_type(8)))  float  v8f;

struct __align__(16) U128 { unsigned int x, y, z, w; };

union B16x16 {                 // one WMMA 16-bit operand (16 bf16 = 32 bytes/lane)
  U128 q[2];
  v16bf v;
  unsigned short h[16];
};

static constexpr int kB  = 4;
static constexpr int kS  = 4096;
static constexpr int kH  = 1024;
static constexpr int kNH = 16;
static constexpr int kHD = 64;
static constexpr int kM  = 256;
static constexpr int kMT = kB * kS;   // 16384 rows (B*S)

// fp32 -> bf16 RNE via the hardware convert (fptrunc); gfx1250 has native
// BF16 VALU support, so this selects a v_cvt instead of a bit-twiddle chain.
__device__ __forceinline__ unsigned short f32_to_bf16(float f) {
  __bf16 b = (__bf16)f;
  return __builtin_bit_cast(unsigned short, b);
}

__device__ __forceinline__ v8f wmma_bf16(const B16x16& a, const B16x16& b, v8f c) {
  // (neg_a, A, neg_b, B, c_mod, C, reuse_a, reuse_b)
  return __builtin_amdgcn_wmma_f32_16x16x32_bf16(false, a.v, false, b.v,
                                                 (short)0, c, false, false);
}

// ---------------------------------------------------------------------------
// Pre-pass: fp32 -> bf16 flat conversion (x).
// ---------------------------------------------------------------------------
__global__ void cvt_bf16(const float* __restrict__ x, unsigned short* __restrict__ y) {
  const size_t i = (size_t)blockIdx.x * blockDim.x + threadIdx.x;
  y[i] = f32_to_bf16(x[i]);
}

// ---------------------------------------------------------------------------
// Pre-pass: pack a row-major [K][N] fp32 matrix into this file's (contiguous-k) WMMA B-register
// layout, tiled [N/16][K/32][lane(32)][slot(16)] bf16:
//   within a 32-K tile: sub<16  -> lane = n%16,      slot = sub
//                       sub>=16 -> lane = 16 + n%16, slot = sub-16
// GEMM-side read: one contiguous 32B load per lane (fully coalesced).
// ---------------------------------------------------------------------------
__global__ void pack_b_kn(const float* __restrict__ W, unsigned short* __restrict__ P,
                          int K, int N) {
  const size_t i = (size_t)blockIdx.x * blockDim.x + threadIdx.x;   // over K*N
  const int k = (int)(i / N), n = (int)(i % N);
  const int kt = k >> 5, sub = k & 31, nt = n >> 4, nn = n & 15;
  const int lane = nn + (sub & 16);
  const int slot = sub & 15;
  P[(((size_t)nt * (K >> 5) + kt) * 32 + lane) * 16 + slot] = f32_to_bf16(W[i]);
}

// Same pack but source is row-major [N][K] (rf is [M][HD] = [n][k]).
__global__ void pack_b_nk(const float* __restrict__ W, unsigned short* __restrict__ P,
                          int K, int N) {
  const size_t i = (size_t)blockIdx.x * blockDim.x + threadIdx.x;   // over N*K
  const int n = (int)(i / K), k = (int)(i % K);
  const int kt = k >> 5, sub = k & 31, nt = n >> 4, nn = n & 15;
  const int lane = nn + (sub & 16);
  const int slot = sub & 15;
  P[(((size_t)nt * (K >> 5) + kt) * 32 + lane) * 16 + slot] = f32_to_bf16(W[i]);
}

// ---------------------------------------------------------------------------
// WMMA GEMM: C[kMT,N] = A[kMT,K](bf16 row-major) * Bpacked + bias.
// 256 threads = 8 waves; each wave computes one 16x16 m-tile x 8 n-tiles
// (block = one m-tile covering all 64 n-tiles of N=1024).
// A operand in contiguous k order (matching the B pack): lanes 0-15 K=0..15,
// lanes 16-31 K=16..31 -> two 16B loads per lane per K-step.
// ---------------------------------------------------------------------------
template <bool OUT_BF16>
__global__ __launch_bounds__(256) void gemm_wmma(
    const unsigned short* __restrict__ A,
    const unsigned short* __restrict__ Bp,
    const float* __restrict__ bias,
    void* __restrict__ Cout, int K, int N) {
  const int lane   = threadIdx.x & 31;
  const int wave   = threadIdx.x >> 5;
  const int mtile  = blockIdx.x;
  const int ntbase = wave * 8;
  const int mrow0  = lane & 15;
  const int half8  = (lane < 16) ? 0 : 8;
  const int KT     = K >> 5;

  const size_t arow = ((size_t)mtile * 16 + mrow0) * K;

  v8f z = {};
  v8f acc[8] = {z, z, z, z, z, z, z, z};

  for (int k0 = 0; k0 < K; k0 += 32) {
    B16x16 a;
    const unsigned short* ap = A + arow + k0 + 2 * half8;
    a.q[0] = *(const U128*)(ap);
    a.q[1] = *(const U128*)(ap + 8);
    if (k0 + 32 < K) __builtin_prefetch(ap + 32, 0, 1);   // global_prefetch_b8
    const int kt = k0 >> 5;
#pragma unroll
    for (int j = 0; j < 8; ++j) {
      B16x16 b;
      b.v = *(const v16bf*)(Bp + (((size_t)(ntbase + j) * KT + kt) * 32 + lane) * 16);
      acc[j] = wmma_bf16(a, b, acc[j]);
    }
  }

#pragma unroll
  for (int j = 0; j < 8; ++j) {
    const int ncol = (ntbase + j) * 16 + (lane & 15);
    const float bv = bias[ncol];
#pragma unroll
    for (int r = 0; r < 8; ++r) {
      const int mrow = mtile * 16 + ((lane < 16) ? r : 8 + r);  // C layout
      const float val = acc[j][r] + bv;
      if (OUT_BF16)
        ((unsigned short*)Cout)[(size_t)mrow * N + ncol] = f32_to_bf16(val);
      else
        ((float*)Cout)[(size_t)mrow * N + ncol] = val;
    }
  }
}

// ---------------------------------------------------------------------------
// Fused per-position Performer attention.
// One wave per (b,s) position; 2 waves (64 threads) per block, 34 KB LDS.
//   qf = relu(q @ rf^T)  [16x256]   (WMMA, staged to LDS in A-reg layout)
//   kf = relu(k @ rf^T)  [16x256]   (WMMA, staged to LDS in B-reg layout)
//   scores = qf @ kf^T * HD^-0.5    (8 WMMA, K=256)
//   softmax over heads (width-16 __shfl_xor inside wave32 halves)
//   out = attn @ v                  (4 WMMA, K padded 16->32 with zeros)
// Per-wave LDS staging needs no barriers: same-wave DS ops are in order.
// ---------------------------------------------------------------------------
__global__ __launch_bounds__(64) void performer_attn(
    const unsigned short* __restrict__ qb,
    const unsigned short* __restrict__ kb,
    const unsigned short* __restrict__ vb,
    const unsigned short* __restrict__ rfP,
    unsigned short* __restrict__ attb) {
  __shared__ __align__(32) unsigned short qfA[2][8][32][16]; // A layout, K=features
  __shared__ __align__(32) unsigned short kfB[2][8][32][16]; // B layout, K=features
  __shared__ __align__(32) unsigned short atA[2][32][16];    // attn A layout, K padded

  const int lane  = threadIdx.x & 31;
  const int w     = threadIdx.x >> 5;
  const size_t pos = (size_t)blockIdx.x * 2 + w;
  const int half8 = (lane < 16) ? 0 : 8;
  const int col   = lane & 15;

  // ---- feature projections for q and k ----
  for (int which = 0; which < 2; ++which) {
    const unsigned short* src  = which ? kb : qb;
    const unsigned short* base = src + pos * (kNH * kHD) + (size_t)col * kHD; // row=head
    for (int t = 0; t < 16; ++t) {          // 16 feature n-tiles of 16
      v8f acc = {};
#pragma unroll
      for (int kt = 0; kt < 2; ++kt) {      // K = HD = 64 -> two 32-steps
        B16x16 a, b;
        const unsigned short* ap = base + kt * 32 + 2 * half8;
        a.q[0] = *(const U128*)ap;
        a.q[1] = *(const U128*)(ap + 8);
        b.v = *(const v16bf*)(rfP + ((t * 2 + kt) * 32 + lane) * 16);
        acc = wmma_bf16(a, b, acc);
      }
      // relu + stage bf16 into LDS in WMMA register layouts
      const int m = t * 16 + col;           // feature index (= K of scores GEMM)
      const int sub = m & 31, ktile = m >> 5;
#pragma unroll
      for (int r = 0; r < 8; ++r) {
        const int row = (lane < 16) ? r : 8 + r;       // C-layout row
        const unsigned short hv = f32_to_bf16(fmaxf(acc[r], 0.0f));
        if (which == 0) {
          // A layout: sub<16 -> (row, sub); sub>=16 -> (16+row, sub-16)
          qfA[w][ktile][row + (sub & 16)][sub & 15] = hv;
        } else {
          // B layout: sub<16 -> (h, sub); sub>=16 -> (16+h, sub-16)
          kfB[w][ktile][row + (sub & 16)][sub & 15] = hv;
        }
      }
    }
  }

  // ---- scores = qf @ kf^T (16x16, K = 256) ----
  v8f sacc = {};
#pragma unroll
  for (int kt = 0; kt < 8; ++kt) {
    B16x16 a, b;
    a.v = *(const v16bf*)&qfA[w][kt][lane][0];
    b.v = *(const v16bf*)&kfB[w][kt][lane][0];
    sacc = wmma_bf16(a, b, sacc);
  }

  // ---- softmax over columns (heads); rows live in 16-lane halves ----
  const float scale = 0.125f;  // 64^-0.5
  float attnv[8];
#pragma unroll
  for (int r = 0; r < 8; ++r) {
    float v = sacc[r] * scale;
    float mx = v;
#pragma unroll
    for (int o = 8; o >= 1; o >>= 1) mx = fmaxf(mx, __shfl_xor(mx, o, 16));
    const float e = __expf(v - mx);
    float sm = e;
#pragma unroll
    for (int o = 8; o >= 1; o >>= 1) sm += __shfl_xor(sm, o, 16);
    attnv[r] = e / sm;
  }

  // ---- stage attn as A tile, K = heads (16) zero-padded to 32 ----
#pragma unroll
  for (int s = 0; s < 4; ++s)
    *(unsigned long long*)&atA[w][lane][s * 4] = 0ull;
#pragma unroll
  for (int r = 0; r < 8; ++r) {
    const int row = (lane < 16) ? r : 8 + r;
    const int h = col;                                   // k index < 16
    atA[w][row][h] = f32_to_bf16(attnv[r]);
  }

  B16x16 aat;
  aat.v = *(const v16bf*)&atA[w][lane][0];

  // ---- out = attn @ v : 4 d-tiles of 16 ----
  const unsigned short* vbase = vb + pos * (kNH * kHD);
#pragma unroll
  for (int dt = 0; dt < 4; ++dt) {
    B16x16 b;   // B[k=head][n=d]: lanes 0-15 heads 0-15, lanes 16-31 pad zeros
#pragma unroll
    for (int s = 0; s < 16; ++s)
      b.h[s] = (lane < 16) ? vbase[(size_t)s * kHD + dt * 16 + lane]
                           : (unsigned short)0;
    v8f oacc = {};
    oacc = wmma_bf16(aat, b, oacc);
#pragma unroll
    for (int r = 0; r < 8; ++r) {
      const int row = (lane < 16) ? r : 8 + r;
      attb[pos * (kNH * kHD) + (size_t)row * kHD + dt * 16 + col] =
          f32_to_bf16(oacc[r]);
    }
  }
}

// ---------------------------------------------------------------------------
// Host launcher
// ---------------------------------------------------------------------------
extern "C" void kernel_launch(void* const* d_in, const int* in_sizes, int n_in,
                              void* d_out, int out_size, void* d_ws, size_t ws_size,
                              hipStream_t stream) {
  (void)in_sizes; (void)n_in; (void)out_size; (void)ws_size;

  const float* x  = (const float*)d_in[0];
  const float* Wq = (const float*)d_in[1];
  const float* bq = (const float*)d_in[2];
  const float* Wk = (const float*)d_in[3];
  const float* bk = (const float*)d_in[4];
  const float* Wv = (const float*)d_in[5];
  const float* bv = (const float*)d_in[6];
  const float* Wo = (const float*)d_in[7];
  const float* bo = (const float*)d_in[8];
  const float* rf = (const float*)d_in[9];
  float* out = (float*)d_out;

  // workspace carve (all offsets 256B-aligned by construction)
  char* w = (char*)d_ws;
  size_t off = 0;
  auto carve = [&](size_t bytes) {
    void* p = w + off;
    off += (bytes + 255) & ~(size_t)255;
    return p;
  };
  const size_t xBytes = (size_t)kMT * kH * sizeof(unsigned short);   // 32 MB
  const size_t wBytes = (size_t)kH * kH * sizeof(unsigned short);    // 2 MB
  unsigned short* xb  = (unsigned short*)carve(xBytes);
  unsigned short* qb  = (unsigned short*)carve(xBytes);
  unsigned short* kbuf= (unsigned short*)carve(xBytes);
  unsigned short* vb  = (unsigned short*)carve(xBytes);
  unsigned short* WqP = (unsigned short*)carve(wBytes);
  unsigned short* WkP = (unsigned short*)carve(wBytes);
  unsigned short* WvP = (unsigned short*)carve(wBytes);
  unsigned short* WoP = (unsigned short*)carve(wBytes);
  unsigned short* rfP = (unsigned short*)carve((size_t)kM * kHD * sizeof(unsigned short));
  unsigned short* attb = xb;   // x retired after projections; reuse for attn output

  // 1) convert x to bf16; pack weights / rf into WMMA B-register layout
  cvt_bf16<<<(kMT * kH) / 256, 256, 0, stream>>>(x, xb);
  const int wBlocks = (kH * kH) / 256;
  pack_b_kn<<<wBlocks, 256, 0, stream>>>(Wq, WqP, kH, kH);
  pack_b_kn<<<wBlocks, 256, 0, stream>>>(Wk, WkP, kH, kH);
  pack_b_kn<<<wBlocks, 256, 0, stream>>>(Wv, WvP, kH, kH);
  pack_b_kn<<<wBlocks, 256, 0, stream>>>(Wo, WoP, kH, kH);
  pack_b_nk<<<(kM * kHD) / 256, 256, 0, stream>>>(rf, rfP, kHD, kM);

  // 2) Q/K/V projections (bf16 out, fp32 accumulate)
  const int gemmBlocks = kMT / 16;   // 1024, one m-tile per block
  gemm_wmma<true><<<gemmBlocks, 256, 0, stream>>>(xb, WqP, bq, qb,   kH, kH);
  gemm_wmma<true><<<gemmBlocks, 256, 0, stream>>>(xb, WkP, bk, kbuf, kH, kH);
  gemm_wmma<true><<<gemmBlocks, 256, 0, stream>>>(xb, WvP, bv, vb,   kH, kH);

  // 3) fused per-position performer attention (one wave per position)
  performer_attn<<<kMT / 2, 64, 0, stream>>>(qb, kbuf, vb, rfP, attb);

  // 4) output projection (fp32 out to d_out)
  gemm_wmma<false><<<gemmBlocks, 256, 0, stream>>>(attb, WoP, bo, out, kH, kH);
}
